// HierMatcher_55697135894806
// MI455X (gfx1250) — matrix operand order repaired and checked by reading, not compiled, not run on hardware
//
#include <hip/hip_runtime.h>
#include <hip/hip_bf16.h>
#include <math.h>
#include <stdint.h>

typedef __attribute__((ext_vector_type(16))) __bf16 v16bf;
typedef __attribute__((ext_vector_type(8)))  float  v8f;
typedef unsigned int uint32x4 __attribute__((ext_vector_type(4)));
typedef int          int32x8  __attribute__((ext_vector_type(8)));
typedef int          int32x4  __attribute__((ext_vector_type(4)));

#define D    768
#define NTOK 256
#define DE   6144
#define BCHUNK 12288   // one B tile: 24*32*16 bf16 elements = 24 KB

// ---------------------------------------------------------------------------
// TDM helper: 1-D DMA of 12288 bf16 (24 KB) from global to LDS.
// D# group0: count=1 | lds_addr | global_addr(57b) | type=2  (ISA 08 §8.3)
// D# group1: data_size=2B, tensor_dim0=tile_dim0=12288, dims1=1 (ISA 08 §8.4)
// ---------------------------------------------------------------------------
__device__ __forceinline__ void tdm_load_24k(const __bf16* gsrc, uint32_t lds_addr)
{
    uint64_t ga = (uint64_t)(uintptr_t)gsrc;
    uint32x4 g0;
    g0.x = 1u;                                                  // count=1, user D#
    g0.y = lds_addr;                                            // LDS byte address
    g0.z = (uint32_t)(ga & 0xFFFFFFFFu);                        // global_addr[31:0]
    g0.w = (uint32_t)((ga >> 32) & 0x01FFFFFFu) | 0x80000000u;  // addr[56:32] | type=2
    int32x8 g1;
    g1[0] = 0x00010000;                          // workgroup_mask=0, data_size=1 (2B)
    g1[1] = (int)((BCHUNK & 0xFFFF) << 16);      // tensor_dim0 lo16
    g1[2] = (int)(1u << 16);                     // tensor_dim0 hi16=0 | tensor_dim1=1
    g1[3] = (int)((BCHUNK & 0xFFFF) << 16);      // tensor_dim1 hi=0 | tile_dim0
    g1[4] = 1;                                   // tile_dim1=1 | tile_dim2=0
    g1[5] = BCHUNK;                              // tensor_dim0_stride lo32
    g1[6] = 0;
    g1[7] = 0;
    int32x4 z4 = {0, 0, 0, 0};
#if __clang_major__ >= 23
    int32x8 z8 = {0, 0, 0, 0, 0, 0, 0, 0};
    __builtin_amdgcn_tensor_load_to_lds(g0, g1, z4, z4, z8, 0);
#else
    __builtin_amdgcn_tensor_load_to_lds(g0, g1, z4, z4, 0);
#endif
}

// ================= Phase 0: pre-swizzle Wn/Wg into WMMA B-fragment layout ====
// Bswz[mat][nt][kc][lane][e] (bf16): lane = n + 16*((k>>3)&1), e = (k&7) + 8*((k>>4)&1)
__global__ __launch_bounds__(256)
void swizzle_weights(const float* __restrict__ Wn, const float* __restrict__ Wg,
                     __bf16* __restrict__ Bswz)
{
    int idx = blockIdx.x * 256 + threadIdx.x;
    if (idx >= 2 * D * D) return;
    int e    = idx & 15;
    int lane = (idx >> 4) & 31;
    int v    = idx >> 9;          // kc + 24*(nt + 48*mat)
    int kc   = v % 24;
    int nt   = (v / 24) % 48;
    int mat  = v / (24 * 48);
    int n = lane & 15;
    int k = kc * 32 + ((lane & 16) ? 8 : 0) + (e & 7) + ((e >= 8) ? 16 : 0);
    const float* W = mat ? Wg : Wn;
    Bswz[idx] = (__bf16)W[(size_t)k * D + nt * 16 + n];
}

// ================= Phase 1: pair score matrix via bf16 WMMA + TDM ===========
// scores[i][j] = sum_d W_lin[d] * highway(|L[i]-R[j]|)[d]  (b_lin: argmax-invariant)
// grid: 1024 blocks = (i in [0,256)) x (j-block in [0,4)); block = 128 = 4 waves.
__global__ __launch_bounds__(128)
void score_kernel(const float* __restrict__ left, const float* __restrict__ right,
                  const __bf16* __restrict__ Bswz,
                  const float* __restrict__ bn, const float* __restrict__ bg,
                  const float* __restrict__ wlin, float* __restrict__ scores)
{
    // A fragments pre-swizzled per wave: [wave][kchunk][lane][16 bf16]  (96 KB)
    __shared__ alignas(32) __bf16 Albuf[4][24][32][16];
    // B fragments for Wn/Wg column block, filled by TDM:                (48 KB)
    __shared__ alignas(32) __bf16 Bbuf[2][24][32][16];
    __shared__ float Lrow[D];                                          //  (3 KB)

    const int tid   = threadIdx.x;
    const int lane  = tid & 31;
    const int w     = tid >> 5;
    const int i     = blockIdx.x >> 2;
    const int jbase = (blockIdx.x & 3) * 64 + w * 16;
    const int half  = lane >> 4;

    for (int d = tid; d < D; d += 128) Lrow[d] = left[i * D + d];
    __syncthreads();

    // ---- stage A: cmp = |L[i]-R[j]| for this wave's 16 pairs, WMMA A-layout ----
    // lane = m + 16*khalf ; element e -> K = e + 8*khalf + (e>=8 ? 8 : 0)
    {
        const int m = lane & 15;
        const float* Rrow = right + (jbase + m) * D;
        for (int kc = 0; kc < 24; ++kc) {
            #pragma unroll
            for (int e = 0; e < 16; ++e) {
                int K = e + 8 * half + ((e >= 8) ? 8 : 0);
                int d = kc * 32 + K;
                Albuf[w][kc][lane][e] = (__bf16)fabsf(Lrow[d] - Rrow[d]);
            }
        }
    }

    float sAcc[8];
    #pragma unroll
    for (int r = 0; r < 8; ++r) sAcc[r] = 0.f;

    const uint32_t ldsB0 = (uint32_t)(uintptr_t)&Bbuf[0][0][0][0];
    const uint32_t ldsB1 = (uint32_t)(uintptr_t)&Bbuf[1][0][0][0];

    for (int nt = 0; nt < 48; ++nt) {
        __syncthreads();   // previous B tile fully consumed
        if (w == 0) {
            // Tensor Data Mover: two 24 KB contiguous B tiles, no VALU staging
            tdm_load_24k(Bswz + (size_t)nt * BCHUNK, ldsB0);
            tdm_load_24k(Bswz + (size_t)(48 + nt) * BCHUNK, ldsB1);
            __builtin_amdgcn_s_wait_tensorcnt((short)0);
        }
        if (nt + 1 < 48 && w == 1) {  // warm L2 for next tiles -> global_prefetch_b8
            __builtin_prefetch(Bswz + (size_t)(nt + 1) * BCHUNK + lane * 384, 0, 1);
            __builtin_prefetch(Bswz + (size_t)(49 + nt) * BCHUNK + lane * 384, 0, 1);
        }
        __syncthreads();   // B tile visible to all waves

        // ---- K loop: 24 x (2 WMMAs) straight from LDS ----
        v8f Cn = {}; v8f Cg = {};
        #pragma unroll 4
        for (int kc = 0; kc < 24; ++kc) {
            v16bf a  = *(const v16bf*)&Albuf[w][kc][lane][0];
            v16bf b0 = *(const v16bf*)&Bbuf[0][kc][lane][0];
            v16bf b1 = *(const v16bf*)&Bbuf[1][kc][lane][0];
            Cn = __builtin_amdgcn_wmma_f32_16x16x32_bf16(false, a, false, b0,
                                                         (short)0, Cn, false, false);
            Cg = __builtin_amdgcn_wmma_f32_16x16x32_bf16(false, a, false, b1,
                                                         (short)0, Cg, false, false);
        }

        // ---- epilogue: highway + fold W_lin reduction; C: col=lane&15, row=r+8*half ----
        int ncol = nt * 16 + (lane & 15);
        float bnv = bn[ncol], bgv = bg[ncol], wl = wlin[ncol];
        int kcn = ncol >> 5, kkn = ncol & 31;
        int khn = (kkn >> 3) & 1;
        int e2  = (kkn & 7) + ((kkn >> 4) << 3);
        #pragma unroll
        for (int r = 0; r < 8; ++r) {
            int m = r + 8 * half;
            float cmpv = (float)Albuf[w][kcn][m + 16 * khn][e2];
            float hv = fmaxf(Cn[r] + bnv, 0.f);
            float gv = 1.f / (1.f + __expf(-(Cg[r] + bgv)));
            float contrib = (hv * gv + (1.f - gv) * cmpv) * wl;
            #pragma unroll
            for (int s = 1; s < 16; s <<= 1)
                contrib += __shfl_xor(contrib, s, 32);
            sAcc[r] += contrib;
        }
    }

    if ((lane & 15) == 0) {
        #pragma unroll
        for (int r = 0; r < 8; ++r) {
            int m = r + 8 * half;
            scores[i * NTOK + jbase + m] = sAcc[r];
        }
    }
}

// ================= Phase 2: row/col argmax (first-max semantics) =================
__global__ __launch_bounds__(256)
void argmax_kernel(const float* __restrict__ scores, int* __restrict__ lidx,
                   int* __restrict__ ridx)
{
    int t = threadIdx.x;
    float best = -INFINITY; int bi = 0;
    for (int j = 0; j < NTOK; ++j) {
        float v = scores[t * NTOK + j];
        if (v > best) { best = v; bi = j; }
    }
    lidx[t] = bi;
    best = -INFINITY; bi = 0;
    for (int i2 = 0; i2 < NTOK; ++i2) {
        float v = scores[i2 * NTOK + t];   // transpose trick: right-side argmax
        if (v > best) { best = v; bi = i2; }
    }
    ridx[t] = bi;
}

// ================= Phase 3: ragged attribute softmax reduction =================
__global__ __launch_bounds__(256)
void attr_kernel(const float* __restrict__ left, const float* __restrict__ right,
                 const float* __restrict__ attrL, const float* __restrict__ attrR,
                 const int* __restrict__ lensL, const int* __restrict__ lensR,
                 const int* __restrict__ lidx, const int* __restrict__ ridx,
                 const float* __restrict__ empty_attr, float* __restrict__ xvec)
{
    __shared__ float wbuf[256];
    __shared__ float red[256];
    int tid  = threadIdx.x;
    int b    = blockIdx.x;
    int side = b >> 2;
    int a    = b & 3;
    const float* emb  = side ? right : left;
    const float* oemb = side ? left  : right;
    const float* attr = side ? attrR : attrL;
    const int*   lens = side ? lensR : lensL;
    const int*   idx  = side ? ridx  : lidx;

    int start = 0;
    for (int q = 0; q < a; ++q) start += lens[q];
    int len = lens[a];
    float* out = xvec + side * (4 * D) + a * D;

    if (len == 0) {
        for (int d = tid; d < D; d += 256) out[d] = empty_attr[d];
        return;
    }
    float sc = -INFINITY;
    if (tid < len) {
        const float* er = emb + (size_t)(start + tid) * D;
        float acc = 0.f;
        for (int d = 0; d < D; ++d) acc = fmaf(er[d], attr[a * D + d], acc);
        sc = acc;
    }
    red[tid] = sc; __syncthreads();
    for (int s = 128; s > 0; s >>= 1) {
        if (tid < s) red[tid] = fmaxf(red[tid], red[tid + s]);
        __syncthreads();
    }
    float m = red[0]; __syncthreads();
    float e = (tid < len) ? __expf(sc - m) : 0.f;
    red[tid] = e; __syncthreads();
    for (int s = 128; s > 0; s >>= 1) {
        if (tid < s) red[tid] += red[tid + s];
        __syncthreads();
    }
    float z = red[0]; __syncthreads();
    wbuf[tid] = e / z; __syncthreads();

    for (int d = tid; d < D; d += 256) {
        float acc = 0.f;
        for (int t = 0; t < len; ++t) {
            int tok = start + t;
            float cmpv = fabsf(emb[(size_t)tok * D + d] - oemb[(size_t)idx[tok] * D + d]);
            acc = fmaf(wbuf[t], cmpv, acc);
        }
        out[d] = acc;
    }
}

// ================= Phase 4: entity highway (bandwidth-bound matvec) =================
__global__ void zero_kernel(float* __restrict__ p, int n)
{
    int t = blockIdx.x * 256 + threadIdx.x;
    if (t < n) p[t] = 0.f;
}

// split-K matvec over two 6144x6144 f32 matrices: 302 MB streamed -> ~13 us @ 23.3 TB/s
__global__ __launch_bounds__(256)
void entity_matvec(const float* __restrict__ x, const float* __restrict__ Wn,
                   const float* __restrict__ Wg, float* __restrict__ hpre,
                   float* __restrict__ gpre)
{
    int col = blockIdx.x * 256 + threadIdx.x;
    int k0  = blockIdx.y * (DE / 8);
    float an = 0.f, ag = 0.f;
    for (int kk = 0; kk < DE / 8; ++kk) {
        int k = k0 + kk;
        float xv = x[k];
        an = fmaf(xv, Wn[(size_t)k * DE + col], an);
        ag = fmaf(xv, Wg[(size_t)k * DE + col], ag);
    }
    atomicAdd(&hpre[col], an);
    atomicAdd(&gpre[col], ag);
}

__global__ __launch_bounds__(256)
void final_kernel(const float* __restrict__ x, const float* __restrict__ hpre,
                  const float* __restrict__ gpre, const float* __restrict__ bn,
                  const float* __restrict__ bg, const float* __restrict__ wlin,
                  const float* __restrict__ blin, float* __restrict__ out)
{
    __shared__ float r0[256], r1[256];
    int tid = threadIdx.x;
    float a0 = 0.f, a1 = 0.f;
    for (int c = tid; c < DE; c += 256) {
        float h  = fmaxf(hpre[c] + bn[c], 0.f);
        float g  = 1.f / (1.f + __expf(-(gpre[c] + bg[c])));
        float hw = h * g + (1.f - g) * x[c];
        a0 = fmaf(hw, wlin[c * 2 + 0], a0);
        a1 = fmaf(hw, wlin[c * 2 + 1], a1);
    }
    r0[tid] = a0; r1[tid] = a1; __syncthreads();
    for (int s = 128; s > 0; s >>= 1) {
        if (tid < s) { r0[tid] += r0[tid + s]; r1[tid] += r1[tid + s]; }
        __syncthreads();
    }
    if (tid == 0) {
        float l0 = r0[0] + blin[0], l1 = r1[0] + blin[1];
        float mx = fmaxf(l0, l1);
        float e0 = __expf(l0 - mx), e1 = __expf(l1 - mx);
        float z  = e0 + e1;
        out[0] = e0 / z;
        out[1] = e1 / z;
    }
}

extern "C" void kernel_launch(void* const* d_in, const int* in_sizes, int n_in,
                              void* d_out, int out_size, void* d_ws, size_t ws_size,
                              hipStream_t stream)
{
    const float* left      = (const float*)d_in[0];
    const float* right     = (const float*)d_in[1];
    const float* Wn_tok    = (const float*)d_in[2];
    const float* bn_tok    = (const float*)d_in[3];
    const float* Wg_tok    = (const float*)d_in[4];
    const float* bg_tok    = (const float*)d_in[5];
    const float* W_lin_tok = (const float*)d_in[6];
    // d_in[7] b_lin_tok: constant shift, argmax-invariant -> unused
    const float* attrL     = (const float*)d_in[8];
    const float* attrR     = (const float*)d_in[9];
    const float* Wn_ent    = (const float*)d_in[10];
    const float* bn_ent    = (const float*)d_in[11];
    const float* Wg_ent    = (const float*)d_in[12];
    const float* bg_ent    = (const float*)d_in[13];
    const float* W_lin_ent = (const float*)d_in[14];
    const float* b_lin_ent = (const float*)d_in[15];
    const float* empty_a   = (const float*)d_in[16];
    const int*   lensL     = (const int*)d_in[17];
    const int*   lensR     = (const int*)d_in[18];
    float* out = (float*)d_out;

    float* ws     = (float*)d_ws;
    float* scores = ws;                       // 65536 f32
    int*   lidx   = (int*)(ws + 65536);       //   256 i32
    int*   ridx   = (int*)(ws + 65792);       //   256 i32
    float* xvec   = ws + 66048;               //  6144 f32
    float* hpre   = ws + 72192;               //  6144 f32
    float* gpre   = ws + 78336;               //  6144 f32 (contiguous with hpre)
    __bf16* Bswz  = (__bf16*)(ws + 84480);    // 2*48*12288 bf16 = 2.36 MB

    swizzle_weights<<<(2 * D * D + 255) / 256, 256, 0, stream>>>(Wn_tok, Wg_tok, Bswz);
    score_kernel<<<1024, 128, 0, stream>>>(left, right, Bswz,
                                           bn_tok, bg_tok, W_lin_tok, scores);
    argmax_kernel<<<1, 256, 0, stream>>>(scores, lidx, ridx);
    attr_kernel<<<8, 256, 0, stream>>>(left, right, attrL, attrR, lensL, lensR,
                                       lidx, ridx, empty_a, xvec);
    zero_kernel<<<(2 * DE + 255) / 256, 256, 0, stream>>>(hpre, 2 * DE);
    entity_matvec<<<dim3(DE / 256, 8), 256, 0, stream>>>(xvec, Wn_ent, Wg_ent,
                                                         hpre, gpre);
    final_kernel<<<1, 256, 0, stream>>>(xvec, hpre, gpre, bn_ent, bg_ent,
                                        W_lin_ent, b_lin_ent, out);
}
